// DumE_27719718928589
// MI455X (gfx1250) — matrix operand order repaired and checked by reading, not compiled, not run on hardware
//
#include <hip/hip_runtime.h>
#include <hip/hip_bf16.h>

// ---------------------------------------------------------------------------
// GPT-2 forward for MI455X (gfx1250, wave32, WMMA + Tensor Data Mover).
// GEMMs: v_wmma_f32_16x16x32_f16. A-tiles staged by TDM (tensor_load_to_lds)
// with D# LDS row padding; B-tiles transposed into LDS manually.
// ---------------------------------------------------------------------------

typedef _Float16 half_t;
typedef __attribute__((ext_vector_type(16))) _Float16 v16h;
typedef __attribute__((ext_vector_type(8)))  float    v8f;
typedef unsigned int u32;
typedef u32 u32x4 __attribute__((ext_vector_type(4)));
typedef int i32x4 __attribute__((ext_vector_type(4)));
typedef int i32x8 __attribute__((ext_vector_type(8)));

#define TB     1024
#define DMODEL 1024
#define NHEAD  16
#define DHEAD  64
#define NLAYER 8
#define VOCAB  50257
#define VPAD   50304
#define DFFN   4096
#define BATCH  2
#define MROWS  (BATCH*TB)

// LDS tile row stride: 32 halves data + 8 halves pad = 80 B (16B-aligned rows,
// matches TDM pad_interval=16dw / pad_amount=4dw exactly)
#define LDST 40

#if defined(__has_builtin)
#if __has_builtin(__builtin_amdgcn_tensor_load_to_lds) && __has_builtin(__builtin_amdgcn_s_wait_tensorcnt)
#define USE_TDM 1
#endif
#endif
#ifndef USE_TDM
#define USE_TDM 0
#endif

union F16Frag { v16h v; unsigned u[8]; };

// A fragment (16x32): lane l%16 = row M; VGPR r holds packed K pair
//   k = (r&3)*2 + (r>>2)*16 + (lane>=16 ? 8 : 0)
__device__ __forceinline__ void load_a_frag(const half_t* lds, int row_base,
                                            int lane, F16Frag& f) {
  const half_t* rp = lds + (row_base + (lane & 15)) * LDST;
  const int hs = (lane >> 4) << 3;
#pragma unroll
  for (int r = 0; r < 8; ++r) {
    int k = ((r & 3) << 1) + ((r >> 2) << 4) + hs;
    f.u[r] = *(const unsigned*)(rp + k);
  }
}

// B fragment (32x16) from transposed tile Bs[n][k]: lane l%16 = col N;
// VGPR r holds K pair k = (r&3)*2 + (r>>2)*16 + (lane>=16 ? 8 : 0)
__device__ __forceinline__ void load_b_frag(const half_t* ldsT, int n_base,
                                            int lane, F16Frag& f) {
  const half_t* rp = ldsT + (n_base + (lane & 15)) * LDST;
  const int hs = (lane >> 4) << 3;
#pragma unroll
  for (int r = 0; r < 8; ++r) {
    int k = ((r & 3) << 1) + ((r >> 2) << 4) + hs;
    f.u[r] = *(const unsigned*)(rp + k);
  }
}

__device__ __forceinline__ v8f wmma_f16(const F16Frag& a, const F16Frag& b, v8f c) {
  return __builtin_amdgcn_wmma_f32_16x16x32_f16(false, a.v, false, b.v,
                                                (short)0, c, false, false);
}

__device__ __forceinline__ unsigned lds_byte_off(const half_t* p) {
  // addrspace(3)->flat cast keeps the LDS offset in the low 32 bits
  return (unsigned)(unsigned long long)(uintptr_t)p;
}

#if USE_TDM
// TDM: load a [rows x 32-half] tile (row stride = stride_elems halves) from
// global into LDS at lds_off, inserting 16B pad per 64B row (LDS stride LDST).
__device__ __forceinline__ void tdm_load_tile(unsigned lds_off,
                                              const half_t* gptr,
                                              unsigned rows,
                                              unsigned stride_elems) {
  unsigned long long ga = (unsigned long long)(uintptr_t)gptr;
  u32x4 g0;
  g0[0] = 1u;                                        // count=1 user descriptor
  g0[1] = lds_off;                                   // lds_addr
  g0[2] = (u32)ga;                                   // global_addr[31:0]
  g0[3] = (u32)((ga >> 32) & 0x01FFFFFFu) | (2u << 30); // addr[56:32] | type=2
  i32x8 g1;
  g1[0] = (int)((1u << 16) | (1u << 20) | (3u << 22) | (3u << 25));
          // data_size=2B, pad_enable, pad_interval=16dw, pad_amount=4dw
  g1[1] = (int)(32u << 16);                          // tensor_dim0 = 32
  g1[2] = (int)((rows & 0xFFFFu) << 16);             // tensor_dim1.lo = rows
  g1[3] = (int)(32u << 16);                          // tile_dim0 = 32
  g1[4] = (int)(rows & 0xFFFFu);                     // tile_dim1 = rows
  g1[5] = (int)stride_elems;                         // tensor_dim0_stride
  g1[6] = 0;
  g1[7] = 0;
  i32x4 z4 = {0, 0, 0, 0};
#if defined(__clang_major__) && (__clang_major__ >= 23)
  i32x8 z8 = {0, 0, 0, 0, 0, 0, 0, 0};
  __builtin_amdgcn_tensor_load_to_lds(g0, g1, z4, z4, z8, 0);
#else
  __builtin_amdgcn_tensor_load_to_lds(g0, g1, z4, z4, 0);
#endif
}
#endif

// ---------------------------------------------------------------------------
// Generic GEMM: out = act(A[M,K]*W[K,N] + bias) + residual.  Block tile
// 128x64, 8 waves as 4(m)x2(n), each wave owns 32x32 C (4 WMMAs / K-step).
// ---------------------------------------------------------------------------
__global__ __launch_bounds__(256) void gemm_wmma_f16(
    const half_t* __restrict__ A, long lda,
    const half_t* __restrict__ W, long ldw,
    const float* __restrict__ bias,
    const float* __restrict__ residual, long ldr,
    float* __restrict__ outF, long ldo,
    half_t* __restrict__ outH, long ldoh,
    int N, int K, int act)
{
  __shared__ half_t As[128 * LDST];
  __shared__ half_t Bs[64 * LDST];

  const int tid  = threadIdx.x;
  const int lane = tid & 31;
  const int wave = tid >> 5;
  const int wm   = wave & 3;          // 0..3 -> 32-row strip
  const int wn   = wave >> 2;         // 0..1 -> 32-col strip
  const long m0  = (long)blockIdx.y * 128;
  const long n0  = (long)blockIdx.x * 64;

  v8f acc[4] = { v8f{}, v8f{}, v8f{}, v8f{} };

  const int bk = tid >> 3, bchunk = tid & 7;       // B: 32 k x (8 x 8 halves)
#if !USE_TDM
  const int arow = tid >> 2, achunk = tid & 3;     // A fallback mapping
#endif

  for (int k0 = 0; k0 < K; k0 += 32) {
#if USE_TDM
    if (wave == 0)
      tdm_load_tile(lds_byte_off(As), A + m0 * lda + k0, 128, (unsigned)lda);
#else
    {
      uint4 av = *(const uint4*)(A + (m0 + arow) * lda + k0 + achunk * 8);
      unsigned* asu = (unsigned*)(As + arow * LDST + achunk * 8);
      asu[0] = av.x; asu[1] = av.y; asu[2] = av.z; asu[3] = av.w;
      uint4 av2 = *(const uint4*)(A + (m0 + 64 + arow) * lda + k0 + achunk * 8);
      unsigned* asu2 = (unsigned*)(As + (64 + arow) * LDST + achunk * 8);
      asu2[0] = av2.x; asu2[1] = av2.y; asu2[2] = av2.z; asu2[3] = av2.w;
    }
#endif
    // B tile transposed: Bs[n][k] = W[k0+k][n0+n]
    uint4 bv = *(const uint4*)(W + (long)(k0 + bk) * ldw + n0 + bchunk * 8);
    union { uint4 q; half_t h[8]; } bu; bu.q = bv;
#pragma unroll
    for (int i = 0; i < 8; ++i)
      Bs[(bchunk * 8 + i) * LDST + bk] = bu.h[i];

    if (k0 + 32 < K)   // global_prefetch_b8 on the weight stream
      __builtin_prefetch(W + (long)(k0 + 32 + bk) * ldw + n0 + bchunk * 8, 0, 3);

#if USE_TDM
    if (wave == 0) __builtin_amdgcn_s_wait_tensorcnt(0);
#endif
    __syncthreads();

    F16Frag af0, af1, bf0, bf1;
    load_a_frag(As, wm * 32,      lane, af0);
    load_a_frag(As, wm * 32 + 16, lane, af1);
    load_b_frag(Bs, wn * 32,      lane, bf0);
    load_b_frag(Bs, wn * 32 + 16, lane, bf1);
    acc[0] = wmma_f16(af0, bf0, acc[0]);
    acc[1] = wmma_f16(af0, bf1, acc[1]);
    acc[2] = wmma_f16(af1, bf0, acc[2]);
    acc[3] = wmma_f16(af1, bf1, acc[3]);
    __syncthreads();
  }

#pragma unroll
  for (int mi = 0; mi < 2; ++mi) {
#pragma unroll
    for (int ni = 0; ni < 2; ++ni) {
      long n = n0 + wn * 32 + ni * 16 + (lane & 15);
      if (n >= N) continue;
      float bi = bias ? bias[n] : 0.f;
      v8f& a = acc[mi * 2 + ni];
#pragma unroll
      for (int r = 0; r < 8; ++r) {
        long m = m0 + wm * 32 + mi * 16 + r + ((lane >> 4) << 3);
        float v = a[r] + bi;
        if (act == 1) v = 0.5f * v * (1.0f + erff(v * 0.70710678118654752f));
        if (residual) v += residual[m * ldr + n];
        if (outF) outF[m * ldo + n] = v;
        if (outH) outH[m * ldoh + n] = (half_t)v;
      }
    }
  }
}

// ---------------------------------------------------------------------------
// Causal scores: S[bh,t,s] = scale * Q[t,:].K[s,:] (s<=t). Both tiles via TDM.
// ---------------------------------------------------------------------------
__global__ __launch_bounds__(256) void attn_scores_wmma(
    const half_t* __restrict__ qkv, float* __restrict__ scores, float scale)
{
  const int kt = blockIdx.x, qt = blockIdx.y, bh = blockIdx.z;
  if (kt > qt) return;
  const int b = bh / NHEAD, h = bh % NHEAD;
  const long ld = 3 * DMODEL;
  const half_t* qbase = qkv + (long)b * TB * ld + h * DHEAD;
  const half_t* kbase = qbase + DMODEL;

  __shared__ half_t As[64 * LDST];
  __shared__ half_t Bs[64 * LDST];

  const int tid = threadIdx.x, lane = tid & 31, wave = tid >> 5;
  const int wm = wave & 3, wn = wave >> 2;
#if !USE_TDM
  const int arow = tid >> 2, achunk = tid & 3;
#endif
  v8f acc[2] = { v8f{}, v8f{} };

  for (int k0 = 0; k0 < DHEAD; k0 += 32) {
#if USE_TDM
    if (wave == 0) {
      tdm_load_tile(lds_byte_off(As), qbase + (long)(qt * 64) * ld + k0, 64, (unsigned)ld);
      tdm_load_tile(lds_byte_off(Bs), kbase + (long)(kt * 64) * ld + k0, 64, (unsigned)ld);
      __builtin_amdgcn_s_wait_tensorcnt(0);
    }
#else
    {
      uint4 av = *(const uint4*)(qbase + (long)(qt * 64 + arow) * ld + k0 + achunk * 8);
      unsigned* asu = (unsigned*)(As + arow * LDST + achunk * 8);
      asu[0] = av.x; asu[1] = av.y; asu[2] = av.z; asu[3] = av.w;
      uint4 bv = *(const uint4*)(kbase + (long)(kt * 64 + arow) * ld + k0 + achunk * 8);
      unsigned* bsu = (unsigned*)(Bs + arow * LDST + achunk * 8);
      bsu[0] = bv.x; bsu[1] = bv.y; bsu[2] = bv.z; bsu[3] = bv.w;
    }
#endif
    __syncthreads();
    F16Frag af, bf0, bf1;
    load_a_frag(As, wm * 16, lane, af);
    load_b_frag(Bs, wn * 32, lane, bf0);
    load_b_frag(Bs, wn * 32 + 16, lane, bf1);
    acc[0] = wmma_f16(af, bf0, acc[0]);
    acc[1] = wmma_f16(af, bf1, acc[1]);
    __syncthreads();
  }

  float* srow = scores + (long)bh * TB * TB;
#pragma unroll
  for (int hlf = 0; hlf < 2; ++hlf) {
    int s = kt * 64 + wn * 32 + hlf * 16 + (lane & 15);
#pragma unroll
    for (int r = 0; r < 8; ++r) {
      int t = qt * 64 + wm * 16 + r + ((lane >> 4) << 3);
      if (s <= t) srow[(long)t * TB + s] = acc[hlf][r] * scale;
    }
  }
}

// ---------------------------------------------------------------------------
// Row softmax over [0..t]; writes f16 probs, zero-filled for s>t.
// ---------------------------------------------------------------------------
__global__ __launch_bounds__(128) void softmax_causal(
    const float* __restrict__ scores, half_t* __restrict__ probs)
{
  const int row = blockIdx.x;
  const int bh = row / TB, t = row % TB, len = t + 1;
  const float* sr = scores + (long)bh * TB * TB + (long)t * TB;
  half_t* pr = probs + (long)bh * TB * TB + (long)t * TB;
  __shared__ float red[128];
  const int tid = threadIdx.x;

  float mx = -3.4e38f;
  for (int i = tid; i < len; i += 128) mx = fmaxf(mx, sr[i]);
  red[tid] = mx; __syncthreads();
  for (int off = 64; off > 0; off >>= 1) {
    if (tid < off) red[tid] = fmaxf(red[tid], red[tid + off]);
    __syncthreads();
  }
  mx = red[0]; __syncthreads();

  float sum = 0.f;
  for (int i = tid; i < len; i += 128) sum += __expf(sr[i] - mx);
  red[tid] = sum; __syncthreads();
  for (int off = 64; off > 0; off >>= 1) {
    if (tid < off) red[tid] += red[tid + off];
    __syncthreads();
  }
  const float inv = 1.0f / red[0];

  for (int i = tid; i < TB; i += 128)
    pr[i] = (i < len) ? (half_t)(__expf(sr[i] - mx) * inv) : (half_t)0.f;
}

// ---------------------------------------------------------------------------
// attn_out = P @ V per (b,h); K-loop truncated at the diagonal. P tile by TDM.
// ---------------------------------------------------------------------------
__global__ __launch_bounds__(256) void attn_pv_wmma(
    const half_t* __restrict__ probs, const half_t* __restrict__ qkv,
    half_t* __restrict__ attnout)
{
  const int qt = blockIdx.y, bh = blockIdx.z;
  const int b = bh / NHEAD, h = bh % NHEAD;
  const long ld = 3 * DMODEL;
  const half_t* P = probs + (long)bh * TB * TB;
  const half_t* vbase = qkv + (long)b * TB * ld + 2 * DMODEL + h * DHEAD;
  half_t* obase = attnout + (long)b * TB * DMODEL + h * DHEAD;

  __shared__ half_t As[64 * LDST];
  __shared__ half_t Bs[64 * LDST];

  const int tid = threadIdx.x, lane = tid & 31, wave = tid >> 5;
  const int wm = wave & 3, wn = wave >> 2;
  const int bs = tid >> 3, bchunk = tid & 7;
#if !USE_TDM
  const int arow = tid >> 2, achunk = tid & 3;
#endif
  v8f acc[2] = { v8f{}, v8f{} };
  const int kmax = (qt + 1) * 64;

  for (int k0 = 0; k0 < kmax; k0 += 32) {
#if USE_TDM
    if (wave == 0)
      tdm_load_tile(lds_byte_off(As), P + (long)(qt * 64) * TB + k0, 64, TB);
#else
    {
      uint4 av = *(const uint4*)(P + (long)(qt * 64 + arow) * TB + k0 + achunk * 8);
      unsigned* asu = (unsigned*)(As + arow * LDST + achunk * 8);
      asu[0] = av.x; asu[1] = av.y; asu[2] = av.z; asu[3] = av.w;
    }
#endif
    // Bs[dh][s] = V[k0+s][dh]
    uint4 bv = *(const uint4*)(vbase + (long)(k0 + bs) * ld + bchunk * 8);
    union { uint4 q; half_t h[8]; } bu; bu.q = bv;
#pragma unroll
    for (int i = 0; i < 8; ++i)
      Bs[(bchunk * 8 + i) * LDST + bs] = bu.h[i];
#if USE_TDM
    if (wave == 0) __builtin_amdgcn_s_wait_tensorcnt(0);
#endif
    __syncthreads();
    F16Frag af, bf0, bf1;
    load_a_frag(As, wm * 16, lane, af);
    load_b_frag(Bs, wn * 32, lane, bf0);
    load_b_frag(Bs, wn * 32 + 16, lane, bf1);
    acc[0] = wmma_f16(af, bf0, acc[0]);
    acc[1] = wmma_f16(af, bf1, acc[1]);
    __syncthreads();
  }

#pragma unroll
  for (int hlf = 0; hlf < 2; ++hlf) {
    int n = wn * 32 + hlf * 16 + (lane & 15);
#pragma unroll
    for (int r = 0; r < 8; ++r) {
      int m = qt * 64 + wm * 16 + r + ((lane >> 4) << 3);
      obase[(long)m * DMODEL + n] = (half_t)acc[hlf][r];
    }
  }
}

// ---------------------------------------------------------------------------
// Elementwise / normalization kernels
// ---------------------------------------------------------------------------
__global__ __launch_bounds__(256) void embed_kernel(
    const int* __restrict__ idx, const float* __restrict__ wte,
    const float* __restrict__ wpe, float* __restrict__ x)
{
  const int row = blockIdx.x;
  const int tok = idx[row], t = row % TB;
  for (int i = threadIdx.x; i < DMODEL; i += 256)
    x[(long)row * DMODEL + i] = wte[(long)tok * DMODEL + i] + wpe[(long)t * DMODEL + i];
}

__global__ __launch_bounds__(256) void layernorm_f16(
    const float* __restrict__ x, const float* __restrict__ w,
    const float* __restrict__ b, half_t* __restrict__ out)
{
  const int row = blockIdx.x;
  const float* xr = x + (long)row * DMODEL;
  __shared__ float red[256];
  const int tid = threadIdx.x;
  float s = 0.f, ss = 0.f;
  for (int i = tid; i < DMODEL; i += 256) { float v = xr[i]; s += v; ss += v * v; }
  red[tid] = s; __syncthreads();
  for (int off = 128; off > 0; off >>= 1) {
    if (tid < off) red[tid] += red[tid + off];
    __syncthreads();
  }
  const float mean = red[0] * (1.0f / DMODEL); __syncthreads();
  red[tid] = ss; __syncthreads();
  for (int off = 128; off > 0; off >>= 1) {
    if (tid < off) red[tid] += red[tid + off];
    __syncthreads();
  }
  const float var = red[0] * (1.0f / DMODEL) - mean * mean;
  const float rstd = rsqrtf(var + 1e-5f);
  for (int i = tid; i < DMODEL; i += 256)
    out[(long)row * DMODEL + i] = (half_t)((xr[i] - mean) * rstd * w[i] + b[i]);
}

__global__ __launch_bounds__(256) void convert_f32_f16(
    const float* __restrict__ src, half_t* __restrict__ dst, long n)
{
  for (long i = (long)blockIdx.x * 256 + threadIdx.x; i < n; i += (long)gridDim.x * 256)
    dst[i] = (half_t)src[i];
}

__global__ __launch_bounds__(256) void transpose_convert_head(
    const float* __restrict__ src, half_t* __restrict__ dst)
{
  const int v = blockIdx.x * 256 + threadIdx.x;
  const int d = blockIdx.y;
  if (v < VPAD)
    dst[(long)d * VPAD + v] = (v < VOCAB) ? (half_t)src[(long)v * DMODEL + d]
                                          : (half_t)0.f;
}

// ---------------------------------------------------------------------------
// Orchestration
// ---------------------------------------------------------------------------
static inline size_t align_up(size_t x) { return (x + 255) & ~(size_t)255; }

extern "C" void kernel_launch(void* const* d_in, const int* in_sizes, int n_in,
                              void* d_out, int out_size, void* d_ws, size_t ws_size,
                              hipStream_t stream)
{
  (void)in_sizes; (void)n_in; (void)out_size; (void)ws_size;
  const int*   idx    = (const int*)  d_in[0];
  const float* wte    = (const float*)d_in[1];
  const float* wpe    = (const float*)d_in[2];
  const float* ln1_w  = (const float*)d_in[3];
  const float* ln1_b  = (const float*)d_in[4];
  const float* qkv_w  = (const float*)d_in[5];
  const float* qkv_b  = (const float*)d_in[6];
  const float* out_w  = (const float*)d_in[7];
  const float* out_b  = (const float*)d_in[8];
  const float* ln2_w  = (const float*)d_in[9];
  const float* ln2_b  = (const float*)d_in[10];
  const float* fc1_w  = (const float*)d_in[11];
  const float* fc1_b  = (const float*)d_in[12];
  const float* fc2_w  = (const float*)d_in[13];
  const float* fc2_b  = (const float*)d_in[14];
  const float* lnf_w  = (const float*)d_in[15];
  const float* lnf_b  = (const float*)d_in[16];
  const float* head_w = (const float*)d_in[17];

  char* ws = (char*)d_ws;
  size_t off = 0;
  float*  x       = (float*)(ws + off); off = align_up(off + (size_t)MROWS * DMODEL * 4);
  half_t* ln16    = (half_t*)(ws + off); off = align_up(off + (size_t)MROWS * DMODEL * 2);
  half_t* qkv16   = (half_t*)(ws + off); off = align_up(off + (size_t)MROWS * 3 * DMODEL * 2);
  half_t* attn16  = (half_t*)(ws + off); off = align_up(off + (size_t)MROWS * DMODEL * 2);
  half_t* h16     = (half_t*)(ws + off); off = align_up(off + (size_t)MROWS * DFFN * 2);
  float*  scores  = (float*)(ws + off); off = align_up(off + (size_t)BATCH * NHEAD * TB * TB * 4);
  half_t* probs16 = (half_t*)(ws + off); off = align_up(off + (size_t)BATCH * NHEAD * TB * TB * 2);
  half_t* w16     = (half_t*)(ws + off); // reused per-GEMM, max D*VPAD halves

  const float scale = 0.125f;  // 1/sqrt(64)
  dim3 blk256(256), blk128(128);

  embed_kernel<<<dim3(MROWS), blk256, 0, stream>>>(idx, wte, wpe, x);

  for (int l = 0; l < NLAYER; ++l) {
    // --- attention ---
    layernorm_f16<<<dim3(MROWS), blk256, 0, stream>>>(
        x, ln1_w + (size_t)l * DMODEL, ln1_b + (size_t)l * DMODEL, ln16);

    convert_f32_f16<<<dim3(4096), blk256, 0, stream>>>(
        qkv_w + (size_t)l * DMODEL * 3 * DMODEL, w16, (long)DMODEL * 3 * DMODEL);
    gemm_wmma_f16<<<dim3(3 * DMODEL / 64, MROWS / 128), blk256, 0, stream>>>(
        ln16, DMODEL, w16, 3 * DMODEL, qkv_b + (size_t)l * 3 * DMODEL,
        nullptr, 0, nullptr, 0, qkv16, 3 * DMODEL, 3 * DMODEL, DMODEL, 0);

    attn_scores_wmma<<<dim3(TB / 64, TB / 64, BATCH * NHEAD), blk256, 0, stream>>>(
        qkv16, scores, scale);
    softmax_causal<<<dim3(BATCH * NHEAD * TB), blk128, 0, stream>>>(scores, probs16);
    attn_pv_wmma<<<dim3(1, TB / 64, BATCH * NHEAD), blk256, 0, stream>>>(
        probs16, qkv16, attn16);

    convert_f32_f16<<<dim3(4096), blk256, 0, stream>>>(
        out_w + (size_t)l * DMODEL * DMODEL, w16, (long)DMODEL * DMODEL);
    gemm_wmma_f16<<<dim3(DMODEL / 64, MROWS / 128), blk256, 0, stream>>>(
        attn16, DMODEL, w16, DMODEL, out_b + (size_t)l * DMODEL,
        x, DMODEL, x, DMODEL, nullptr, 0, DMODEL, DMODEL, 0);

    // --- MLP ---
    layernorm_f16<<<dim3(MROWS), blk256, 0, stream>>>(
        x, ln2_w + (size_t)l * DMODEL, ln2_b + (size_t)l * DMODEL, ln16);

    convert_f32_f16<<<dim3(4096), blk256, 0, stream>>>(
        fc1_w + (size_t)l * DMODEL * DFFN, w16, (long)DMODEL * DFFN);
    gemm_wmma_f16<<<dim3(DFFN / 64, MROWS / 128), blk256, 0, stream>>>(
        ln16, DMODEL, w16, DFFN, fc1_b + (size_t)l * DFFN,
        nullptr, 0, nullptr, 0, h16, DFFN, DFFN, DMODEL, 1 /*gelu*/);

    convert_f32_f16<<<dim3(4096), blk256, 0, stream>>>(
        fc2_w + (size_t)l * DFFN * DMODEL, w16, (long)DFFN * DMODEL);
    gemm_wmma_f16<<<dim3(DMODEL / 64, MROWS / 128), blk256, 0, stream>>>(
        h16, DFFN, w16, DMODEL, fc2_b + (size_t)l * DMODEL,
        x, DMODEL, x, DMODEL, nullptr, 0, DMODEL, DFFN, 0);
  }

  // final LN + LM head
  layernorm_f16<<<dim3(MROWS), blk256, 0, stream>>>(x, lnf_w, lnf_b, ln16);
  transpose_convert_head<<<dim3(VPAD / 256, DMODEL), blk256, 0, stream>>>(head_w, w16);
  gemm_wmma_f16<<<dim3(VPAD / 64, MROWS / 128), blk256, 0, stream>>>(
      ln16, DMODEL, w16, VPAD, nullptr,
      nullptr, 0, (float*)d_out, VOCAB, nullptr, 0, VOCAB, DMODEL, 0);
}
